// LAUniGAT_21131239096595
// MI455X (gfx1250) — compile-verified, matrix operand order checked by reading, not run on hardware
//
#include <hip/hip_runtime.h>
#include <hip/hip_bf16.h>

// ---------------------------------------------------------------------------
// UniGNN-GAT (hypergraph GAT), 2 layers, MI455X (gfx1250, wave32, WMMA).
// GEMMs: bf16 operands pre-staged in global (A row-major, W transposed +
// zero-padded to 64 cols), fragments loaded as 2x b128 per operand straight
// from global per the ISA 16-bit fragment layout -> no LDS, no divergence,
// v_wmma_f32_16x16x32_bf16 with fp32 accumulate.
// Segment reductions: non-returning global_atomic_add_f32; segment-max via
// order-preserving uint encoding + integer atomicMax.
// ---------------------------------------------------------------------------

typedef __attribute__((ext_vector_type(16))) __bf16 v16bf;
typedef __attribute__((ext_vector_type(8)))  __bf16 v8bf;
typedef __attribute__((ext_vector_type(4)))  __bf16 v4bf;
typedef __attribute__((ext_vector_type(8)))  float  v8f;

#define NEG_SLOPE 0.2f
#define ORD_NEG_MAX 0x00800000u  // ord-encoding of -FLT_MAX

__device__ __forceinline__ void atomAddF(float* p, float v) {
  unsafeAtomicAdd(p, v);  // -> global_atomic_add_f32 (no return)
}

__device__ __forceinline__ unsigned f32_ord(float f) {
  unsigned u = __float_as_uint(f);
  return (u & 0x80000000u) ? ~u : (u | 0x80000000u);
}
__device__ __forceinline__ float ord_f32(unsigned u) {
  return (u & 0x80000000u) ? __uint_as_float(u & 0x7fffffffu)
                           : __uint_as_float(~u);
}

__device__ __forceinline__ float leaky(float s) {
  return s >= 0.f ? s : NEG_SLOPE * s;
}

// ------------------------------- fills / converts ---------------------------
__global__ void fill_f32_kernel(float* __restrict__ p, float v, long long n) {
  long long i = (long long)blockIdx.x * blockDim.x + threadIdx.x;
  if (i < n) p[i] = v;
}
__global__ void fill_u32_kernel(unsigned* __restrict__ p, unsigned v, long long n) {
  long long i = (long long)blockIdx.x * blockDim.x + threadIdx.x;
  if (i < n) p[i] = v;
}

// fp32 -> bf16, 4 elems/thread (n must be a multiple of 4; ours are)
__global__ void cvt_bf16_kernel(const float* __restrict__ src, __bf16* __restrict__ dst,
                                long long n) {
  long long i = ((long long)blockIdx.x * blockDim.x + threadIdx.x) * 4;
  if (i >= n) return;
  float4 v = *(const float4*)(src + i);
  v4bf o = {(__bf16)v.x, (__bf16)v.y, (__bf16)v.z, (__bf16)v.w};
  *(v4bf*)(dst + i) = o;
}

// WT[n][k] = bf16(W[k][n]) for n<Nc else 0; WT is [64][K] row-major.
__global__ void wt_cvt_kernel(const float* __restrict__ W, __bf16* __restrict__ WT,
                              int K, int Nc) {
  int gid = blockIdx.x * blockDim.x + threadIdx.x;
  int n = gid / K, k = gid - n * K;
  if (n >= 64) return;
  float v = (n < Nc) ? W[(size_t)k * Nc + n] : 0.f;
  WT[(size_t)n * K + k] = (__bf16)v;
}

// ------------------------------- WMMA GEMM ---------------------------------
// out[M,Nc] = A[M,K](bf16) @ W + bias, with WT = 64xK bf16 (zero-padded).
// 256 threads = 8 waves; wave (rt,ct) owns the 16x16 tile at rows
// blk*32+rt*16, cols ct*16. Per ISA 16-bit fragment layout, lane (m=lane%16)
// needs K elems {kh..kh+7} and {kh+16..kh+23} (kh = lane&16 ? 8 : 0):
// exactly two b128 loads per operand per 32-wide K step.
__global__ __launch_bounds__(256)
void gemm_bf16_wmma_kernel(const __bf16* __restrict__ A, const __bf16* __restrict__ WT,
                           const float* __restrict__ bias, float* __restrict__ out,
                           int M, int K, int Nc) {
  const int tid  = threadIdx.x;
  const int lane = tid & 31;
  const int wave = tid >> 5;
  const int rt   = wave >> 2;  // 0..1
  const int ct   = wave & 3;   // 0..3
  const int row  = blockIdx.x * 32 + rt * 16 + (lane & 15);
  const int rowc = row < M ? row : M - 1;  // clamp loads; stores are guarded
  const int col  = ct * 16 + (lane & 15);  // < 64 always; WT is padded
  const int kh   = (lane & 16) ? 8 : 0;

  const __bf16* arow = A + (size_t)rowc * K + kh;
  const __bf16* brow = WT + (size_t)col * K + kh;

  v8f acc = {};
  for (int kk = 0; kk < K; kk += 32) {
    __builtin_prefetch(arow + kk + 64, 0, 0);
    v8bf a0 = *(const v8bf*)(arow + kk);
    v8bf a1 = *(const v8bf*)(arow + kk + 16);
    v8bf b0 = *(const v8bf*)(brow + kk);
    v8bf b1 = *(const v8bf*)(brow + kk + 16);
    v16bf a = __builtin_shufflevector(a0, a1, 0, 1, 2, 3, 4, 5, 6, 7,
                                      8, 9, 10, 11, 12, 13, 14, 15);
    v16bf b = __builtin_shufflevector(b0, b1, 0, 1, 2, 3, 4, 5, 6, 7,
                                      8, 9, 10, 11, 12, 13, 14, 15);
    acc = __builtin_amdgcn_wmma_f32_16x16x32_bf16(
        false, a, false, b, (short)0, acc, false, false);
  }

  const float bv = (col < Nc) ? bias[col] : 0.f;
#pragma unroll
  for (int r = 0; r < 8; r++) {
    int rg = blockIdx.x * 32 + rt * 16 + r + ((lane >> 4) << 3);
    if (rg < M && col < Nc) out[(size_t)rg * Nc + col] = acc[r] + bv;
  }
}

// --------------------------- graph kernels ---------------------------------
__global__ void count_deg_kernel(const int* __restrict__ pe, float* __restrict__ deg, int P) {
  int p = blockIdx.x * blockDim.x + threadIdx.x;
  if (p >= P) return;
  __builtin_prefetch(pe + p + 4096, 0, 0);
  atomAddF(&deg[pe[p]], 1.0f);
}

__global__ void invert_deg_kernel(float* __restrict__ deg, int E) {
  int e = blockIdx.x * blockDim.x + threadIdx.x;
  if (e < E) deg[e] = 1.0f / fmaxf(deg[e], 1.0f);
}

// Y[pe[p]][c] += Xp[pv[p]][c]   (P*d threads, c fastest -> coalesced)
__global__ void v2e_scatter_kernel(const float* __restrict__ Xp, const int* __restrict__ pe,
                                   const int* __restrict__ pv, float* __restrict__ Y,
                                   int P, int d) {
  int gid = blockIdx.x * blockDim.x + threadIdx.x;
  int p = gid / d, c = gid - p * d;
  if (p >= P) return;
  int v = pv[p], e = pe[p];
  atomAddF(&Y[(size_t)e * d + c], Xp[(size_t)v * d + c]);
}

__global__ void scale_rows_kernel(float* __restrict__ Y, const float* __restrict__ inv,
                                  int R, int d) {
  int gid = blockIdx.x * blockDim.x + threadIdx.x;
  int r = gid / d;
  if (r >= R) return;
  Y[gid] *= inv[r];
}

__global__ void row_dot_kernel(const float* __restrict__ Mrows, const float* __restrict__ vec,
                               float* __restrict__ outv, int R, int d) {
  int r = blockIdx.x * blockDim.x + threadIdx.x;
  if (r >= R) return;
  const float* row = Mrows + (size_t)r * d;
  float s = 0.f;
  for (int c = 0; c < d; c++) s = fmaf(row[c], vec[c], s);
  outv[r] = s;
}

__global__ void pair_max_kernel(const int* __restrict__ pe, const int* __restrict__ pv,
                                const float* __restrict__ se, const float* __restrict__ sv,
                                unsigned* __restrict__ menc, int P) {
  int p = blockIdx.x * blockDim.x + threadIdx.x;
  if (p >= P) return;
  __builtin_prefetch(pe + p + 4096, 0, 0);
  __builtin_prefetch(pv + p + 4096, 0, 0);
  float s = leaky(se[pe[p]] + sv[pv[p]]);
  atomicMax(&menc[pv[p]], f32_ord(s));
}

__global__ void pair_exp_kernel(const int* __restrict__ pe, const int* __restrict__ pv,
                                const float* __restrict__ se, const float* __restrict__ sv,
                                const unsigned* __restrict__ menc, float* __restrict__ ex,
                                float* __restrict__ den, int P) {
  int p = blockIdx.x * blockDim.x + threadIdx.x;
  if (p >= P) return;
  int v = pv[p];
  float s = leaky(se[pe[p]] + sv[v]);
  float e = __expf(s - ord_f32(menc[v]));
  ex[p] = e;
  atomAddF(&den[v], e);
}

// out[pv[p]*ostride + ocol + c] += (ex[p]/den[pv[p]]) * Y[pe[p]*d + c]
__global__ void e2v_scatter_kernel(const float* __restrict__ Y, const int* __restrict__ pe,
                                   const int* __restrict__ pv, const float* __restrict__ ex,
                                   const float* __restrict__ den, float* __restrict__ out,
                                   int P, int d, int ostride, int ocol) {
  int gid = blockIdx.x * blockDim.x + threadIdx.x;
  int p = gid / d, c = gid - p * d;
  if (p >= P) return;
  int v = pv[p], e = pe[p];
  float alpha = ex[p] / den[v];
  atomAddF(&out[(size_t)v * ostride + ocol + c], alpha * Y[(size_t)e * d + c]);
}

__global__ void elu_slice_kernel(float* __restrict__ h, int R, int d, int stride, int off) {
  int gid = blockIdx.x * blockDim.x + threadIdx.x;
  int r = gid / d, c = gid - r * d;
  if (r >= R) return;
  float* p = &h[(size_t)r * stride + off + c];
  float x = *p;
  *p = x > 0.f ? x : (__expf(x) - 1.f);
}

// ------------------------------- host --------------------------------------
static inline unsigned nblk(long long n, int b) { return (unsigned)((n + b - 1) / b); }

extern "C" void kernel_launch(void* const* d_in, const int* in_sizes, int n_in,
                              void* d_out, int out_size, void* d_ws, size_t ws_size,
                              hipStream_t stream) {
  const int din = 128, dh = 64, H = 4, C = 40;
  const int E = 20000;                 // num_e (device scalar; statically known)
  const int P = in_sizes[10];          // incidence pairs
  const int N = in_sizes[0] / din;     // nodes
  const int D2 = 2 * H * dh;           // 512 concat feature dim

  const float* xin[2] = {(const float*)d_in[0], (const float*)d_in[1]};
  const float* W1  = (const float*)d_in[2];
  const float* b1  = (const float*)d_in[3];
  const float* ae1 = (const float*)d_in[4];
  const float* av1 = (const float*)d_in[5];
  const float* W2  = (const float*)d_in[6];
  const float* b2  = (const float*)d_in[7];
  const float* ae2 = (const float*)d_in[8];
  const float* av2 = (const float*)d_in[9];
  const int* pe = (const int*)d_in[10];
  const int* pv = (const int*)d_in[11];
  float* out = (float*)d_out;

  // Workspace carve-out (~396 MB), 256B aligned slices.
  char* wsp = (char*)d_ws;
  auto carve = [&](size_t bytes) {
    char* p = wsp;
    wsp += (bytes + 255) & ~(size_t)255;
    return (void*)p;
  };
  float*    h    = (float*)carve((size_t)N * D2 * 4);   // concat features [N,512] f32
  __bf16*   hb   = (__bf16*)carve((size_t)N * D2 * 2);  // h in bf16 for GEMM2
  __bf16*   xb0  = (__bf16*)carve((size_t)N * din * 2); // x0 bf16
  __bf16*   xb1  = (__bf16*)carve((size_t)N * din * 2); // x1 bf16
  __bf16*   WT   = (__bf16*)carve((size_t)64 * D2 * 2); // transposed weights (padded)
  float*    Xp   = (float*)carve((size_t)N * dh * 4);   // per-combo Xp (L2 reuse: [N,C])
  float*    Y    = (float*)carve((size_t)E * dh * 4);   // per-combo Y  (L2 reuse: [E,C])
  float*    degi = (float*)carve((size_t)E * 4);        // 1/max(deg_e,1)
  float*    se   = (float*)carve((size_t)E * 4);
  float*    sv   = (float*)carve((size_t)N * 4);
  unsigned* menc = (unsigned*)carve((size_t)N * 4);
  float*    den  = (float*)carve((size_t)N * 4);
  float*    ex   = (float*)carve((size_t)P * 4);
  const __bf16* xb[2] = {xb0, xb1};
  (void)ws_size; (void)n_in; (void)out_size;

  const int B = 256;

  // Edge degrees (shared by every layer/head).
  fill_f32_kernel<<<nblk(E, B), B, 0, stream>>>(degi, 0.f, E);
  count_deg_kernel<<<nblk(P, B), B, 0, stream>>>(pe, degi, P);
  invert_deg_kernel<<<nblk(E, B), B, 0, stream>>>(degi, E);

  // bf16 copies of the GEMM inputs.
  cvt_bf16_kernel<<<nblk((long long)N * din / 4, B), B, 0, stream>>>(
      xin[0], xb0, (long long)N * din);
  cvt_bf16_kernel<<<nblk((long long)N * din / 4, B), B, 0, stream>>>(
      xin[1], xb1, (long long)N * din);

  // h accumulates all 8 disjoint column slices; zero once.
  fill_f32_kernel<<<nblk((long long)N * D2, B), B, 0, stream>>>(h, 0.f, (long long)N * D2);

  // ---- Layer 1: 2 inputs x 4 heads ----
  for (int inp = 0; inp < 2; inp++) {
    for (int hd = 0; hd < H; hd++) {
      const float* Wh  = W1  + (size_t)hd * din * dh;
      const float* bh  = b1  + (size_t)hd * dh;
      const float* aeh = ae1 + (size_t)hd * dh;
      const float* avh = av1 + (size_t)hd * dh;
      const int off = inp * H * dh + hd * dh;

      wt_cvt_kernel<<<nblk(64 * din, B), B, 0, stream>>>(Wh, WT, din, dh);
      gemm_bf16_wmma_kernel<<<nblk(N, 32), B, 0, stream>>>(xb[inp], WT, bh, Xp, N, din, dh);

      fill_f32_kernel<<<nblk((long long)E * dh, B), B, 0, stream>>>(Y, 0.f, (long long)E * dh);
      v2e_scatter_kernel<<<nblk((long long)P * dh, B), B, 0, stream>>>(Xp, pe, pv, Y, P, dh);
      scale_rows_kernel<<<nblk((long long)E * dh, B), B, 0, stream>>>(Y, degi, E, dh);

      row_dot_kernel<<<nblk(E, B), B, 0, stream>>>(Y, aeh, se, E, dh);
      row_dot_kernel<<<nblk(N, B), B, 0, stream>>>(Xp, avh, sv, N, dh);

      fill_u32_kernel<<<nblk(N, B), B, 0, stream>>>(menc, ORD_NEG_MAX, N);
      fill_f32_kernel<<<nblk(N, B), B, 0, stream>>>(den, 0.f, N);
      pair_max_kernel<<<nblk(P, B), B, 0, stream>>>(pe, pv, se, sv, menc, P);
      pair_exp_kernel<<<nblk(P, B), B, 0, stream>>>(pe, pv, se, sv, menc, ex, den, P);

      e2v_scatter_kernel<<<nblk((long long)P * dh, B), B, 0, stream>>>(
          Y, pe, pv, ex, den, h, P, dh, D2, off);
      elu_slice_kernel<<<nblk((long long)N * dh, B), B, 0, stream>>>(h, N, dh, D2, off);
    }
  }

  // ---- Layer 2: [N,512] @ [512,40] + UniGAT block into d_out ----
  float* Xp2 = Xp;  // [N,C]
  float* Y2  = Y;   // [E,C]

  cvt_bf16_kernel<<<nblk((long long)N * D2 / 4, B), B, 0, stream>>>(h, hb, (long long)N * D2);
  wt_cvt_kernel<<<nblk(64 * D2, B), B, 0, stream>>>(W2, WT, D2, C);
  gemm_bf16_wmma_kernel<<<nblk(N, 32), B, 0, stream>>>(hb, WT, b2, Xp2, N, D2, C);

  fill_f32_kernel<<<nblk((long long)E * C, B), B, 0, stream>>>(Y2, 0.f, (long long)E * C);
  v2e_scatter_kernel<<<nblk((long long)P * C, B), B, 0, stream>>>(Xp2, pe, pv, Y2, P, C);
  scale_rows_kernel<<<nblk((long long)E * C, B), B, 0, stream>>>(Y2, degi, E, C);

  row_dot_kernel<<<nblk(E, B), B, 0, stream>>>(Y2, ae2, se, E, C);
  row_dot_kernel<<<nblk(N, B), B, 0, stream>>>(Xp2, av2, sv, N, C);

  fill_u32_kernel<<<nblk(N, B), B, 0, stream>>>(menc, ORD_NEG_MAX, N);
  fill_f32_kernel<<<nblk(N, B), B, 0, stream>>>(den, 0.f, N);
  pair_max_kernel<<<nblk(P, B), B, 0, stream>>>(pe, pv, se, sv, menc, P);
  pair_exp_kernel<<<nblk(P, B), B, 0, stream>>>(pe, pv, se, sv, menc, ex, den, P);

  fill_f32_kernel<<<nblk((long long)N * C, B), B, 0, stream>>>(out, 0.f, (long long)N * C);
  e2v_scatter_kernel<<<nblk((long long)P * C, B), B, 0, stream>>>(
      Y2, pe, pv, ex, den, out, P, C, C, 0);
  elu_slice_kernel<<<nblk((long long)N * C, B), B, 0, stream>>>(out, N, C, C, 0);
}
